// ResidualSSM_30021821399781
// MI455X (gfx1250) — compile-verified, matrix operand order checked by reading, not run on hardware
//
#include <hip/hip_runtime.h>
#include <hip/hip_bf16.h>
#include <math.h>

// ---------------------------------------------------------------------------
// Problem constants (match the JAX reference)
// ---------------------------------------------------------------------------
#define B_SZ      4096
#define N_WIN     12
#define D_INPUT   1536
#define D_SCORES  234
#define D_IN_TOT  1770          // D_INPUT + D_SCORES
#define K_IN_PAD  1792          // padded to multiple of 32
#define D_MODEL   128
#define D_STATE   16
#define N_CLASSES 234
#define N_CLS_PAD 240
#define M_ROWS    (B_SZ * N_WIN)    // 49152

typedef _Float16 __attribute__((ext_vector_type(16))) v16h;
typedef float    __attribute__((ext_vector_type(8)))  v8f;

enum { EPI_F32 = 1, EPI_F16 = 2, EPI_SOFTPLUS = 4 };

// ---------------------------------------------------------------------------
// Small math helpers
// ---------------------------------------------------------------------------
__device__ __forceinline__ float gelu_exact(float x) {
    return 0.5f * x * (1.0f + erff(x * 0.70710678118654752f));
}
__device__ __forceinline__ float softplus_f(float x) {
    return (x > 20.0f) ? x : log1pf(expf(x));
}
__device__ __forceinline__ float silu_f(float x) {
    return x / (1.0f + expf(-x));
}
__device__ __forceinline__ float wave_sum32(float v) {
#pragma unroll
    for (int off = 16; off > 0; off >>= 1) v += __shfl_xor(v, off, 32);
    return v;
}

// ---------------------------------------------------------------------------
// Weight prep: src f32 [K x N] row-major -> dst f16 [Npad x Kpad] (transposed,
// zero padded).  Makes B-matrix fragments contiguous 16B reads.
// ---------------------------------------------------------------------------
__global__ void k_transpose_f16(const float* __restrict__ src, _Float16* __restrict__ dst,
                                int K, int N, int Kpad, int Npad) {
    int e = blockIdx.x * 256 + threadIdx.x;
    if (e >= Kpad * Npad) return;
    int n = e / Kpad;
    int k = e - n * Kpad;
    float v = (k < K && n < N) ? src[(size_t)k * N + n] : 0.0f;
    dst[e] = (_Float16)v;
}

// ---------------------------------------------------------------------------
// Generic WMMA GEMM:  C[M x N] = A[M x K] * B[K x N] (+bias) with epilogues.
//   Block: 128 threads = 4 waves; each wave owns 16 rows x (NT*16) cols.
//   A source: ASRC==0 -> f16 matrix (lda); ASRC==1 -> on-the-fly concat of
//   fp32 emb (1536) + scores (234), zero-padded to 1792, converted to f16.
//   Bt is pre-transposed f16 [Npad x Kpad] (Bt[n*ldb + k]).
//   Fragment layouts follow CDNA5 ISA 7.12.2 (wave32).
// ---------------------------------------------------------------------------
template <int NT, int ASRC, int EPI>
__global__ __launch_bounds__(128) void k_gemm_wmma(
    const _Float16* __restrict__ A,
    const float* __restrict__ A1, const float* __restrict__ A2,
    const _Float16* __restrict__ Bt,
    const float* __restrict__ bias,
    float* __restrict__ outF32, _Float16* __restrict__ outF16,
    int M, int K, int lda, int ldb, int ldc, int realN)
{
    __shared__ _Float16 As[64 * 32];
    __shared__ _Float16 Bs[NT * 16 * 32];

    const int tid  = threadIdx.x;
    const int lane = tid & 31;
    const int wave = tid >> 5;
    const int half = lane >> 4;        // 0: lanes 0-15, 1: lanes 16-31
    const int rc   = lane & 15;        // row (A) / col (B/C) within tile
    const int kb   = half * 8;         // K sub-offset per ISA layout

    const int rowBlk = blockIdx.x * 64;
    const int n0     = blockIdx.y * (NT * 16);

    v8f acc[NT] = {};

    for (int k0 = 0; k0 < K; k0 += 32) {
        if (k0) __syncthreads();
        // ---- stage A tile: 64 rows x 32 k (f32 -> f16 convert for concat) --
        for (int e = tid; e < 64 * 32; e += 128) {
            int r = e >> 5, k = e & 31;
            int grow = rowBlk + r;
            _Float16 v;
            if (ASRC == 0) {
                v = A[(size_t)grow * lda + (k0 + k)];
            } else {
                int gk = k0 + k;
                float f;
                if (gk < D_INPUT)         f = A1[(size_t)grow * D_INPUT + gk];
                else if (gk < D_IN_TOT)   f = A2[(size_t)grow * D_SCORES + (gk - D_INPUT)];
                else                      f = 0.0f;
                v = (_Float16)f;
            }
            As[e] = v;
        }
        // ---- stage B tile: NT*16 cols x 32 k -------------------------------
        for (int e = tid; e < NT * 16 * 32; e += 128) {
            int n = e >> 5, k = e & 31;
            Bs[e] = Bt[(size_t)(n0 + n) * ldb + (k0 + k)];
        }
        // prefetch next B chunk into GL2
        if (k0 + 32 < K)
            __builtin_prefetch(&Bt[(size_t)n0 * ldb + (k0 + 32)], 0, 1);
        __syncthreads();

        // ---- A fragment for this wave's 16 rows ----------------------------
        const _Float16* arow = &As[(wave * 16 + rc) * 32];
        v16h afrag;
#pragma unroll
        for (int j = 0; j < 8; ++j) {
            afrag[j]     = arow[kb + j];
            afrag[8 + j] = arow[kb + 16 + j];
        }
        // ---- NT WMMAs ------------------------------------------------------
#pragma unroll
        for (int nt = 0; nt < NT; ++nt) {
            const _Float16* brow = &Bs[(nt * 16 + rc) * 32];
            v16h bfrag;
#pragma unroll
            for (int j = 0; j < 8; ++j) {
                bfrag[j]     = brow[kb + j];
                bfrag[8 + j] = brow[kb + 16 + j];
            }
            acc[nt] = __builtin_amdgcn_wmma_f32_16x16x32_f16(
                false, afrag, false, bfrag, (short)0, acc[nt], false, false);
        }
    }

    // ---- epilogue: C layout (ISA): VGPR i -> row = half*8 + i, col = rc ----
    const int rowBase = rowBlk + wave * 16 + half * 8;
#pragma unroll
    for (int nt = 0; nt < NT; ++nt) {
        int col = n0 + nt * 16 + rc;
        if (col >= realN) continue;
        float bv = bias ? bias[col] : 0.0f;
#pragma unroll
        for (int i = 0; i < 8; ++i) {
            int row = rowBase + i;
            float v = acc[nt][i] + bv;
            if (EPI & EPI_SOFTPLUS) v = softplus_f(v);
            size_t o = (size_t)row * ldc + col;
            if (EPI & EPI_F32) outF32[o] = v;
            if (EPI & EPI_F16) outF16[o] = (_Float16)v;
        }
    }
}

// ---------------------------------------------------------------------------
// Meta path: meta_vec[b,d] = concat(site_emb[site], hour_emb[hour]) @ meta_w + b
// ---------------------------------------------------------------------------
__global__ void k_meta(const int* __restrict__ site_ids, const int* __restrict__ hours,
                       const float* __restrict__ site_emb, const float* __restrict__ hour_emb,
                       const float* __restrict__ meta_w, const float* __restrict__ meta_b,
                       float* __restrict__ meta_vec)
{
    int idx = blockIdx.x * 256 + threadIdx.x;
    if (idx >= B_SZ * D_MODEL) return;
    int b = idx >> 7, d = idx & 127;
    int s = site_ids[b], h = hours[b];
    float acc = meta_b[d];
#pragma unroll
    for (int j = 0; j < 8; ++j) acc += site_emb[s * 8 + j] * meta_w[j * D_MODEL + d];
#pragma unroll
    for (int j = 0; j < 8; ++j) acc += hour_emb[h * 8 + j] * meta_w[(8 + j) * D_MODEL + d];
    meta_vec[idx] = acc;
}

// ---------------------------------------------------------------------------
// Epilogue after input GEMM: h0 = gelu(LN(hpre)) + pos_enc[t] + meta_vec[b]
// One wave32 per row; lane covers cols {lane, lane+32, lane+64, lane+96}.
// ---------------------------------------------------------------------------
__global__ __launch_bounds__(256) void k_epi_input(
    const float* __restrict__ hpre,
    const float* __restrict__ ln_g, const float* __restrict__ ln_b,
    const float* __restrict__ pos,  const float* __restrict__ meta_vec,
    float* __restrict__ h0f32, _Float16* __restrict__ h0f16)
{
    int row  = blockIdx.x * 8 + (threadIdx.x >> 5);
    int lane = threadIdx.x & 31;
    int b = row / N_WIN, t = row - b * N_WIN;
    const float* hr = hpre + (size_t)row * D_MODEL;

    float v[4], sum = 0.0f;
#pragma unroll
    for (int j = 0; j < 4; ++j) { v[j] = hr[lane + 32 * j]; sum += v[j]; }
    sum = wave_sum32(sum);
    float mu = sum * (1.0f / 128.0f);
    float sq = 0.0f;
#pragma unroll
    for (int j = 0; j < 4; ++j) { float d = v[j] - mu; sq += d * d; }
    sq = wave_sum32(sq);
    float rstd = rsqrtf(sq * (1.0f / 128.0f) + 1e-5f);

#pragma unroll
    for (int j = 0; j < 4; ++j) {
        int c = lane + 32 * j;
        float y = (v[j] - mu) * rstd * ln_g[c] + ln_b[c];
        y = gelu_exact(y);
        y += pos[t * D_MODEL + c] + meta_vec[(size_t)b * D_MODEL + c];
        size_t o = (size_t)row * D_MODEL + c;
        h0f32[o] = y;
        h0f16[o] = (_Float16)y;
    }
}

// ---------------------------------------------------------------------------
// Depthwise causal conv1d (D_CONV=4), with time-reversal flag for bwd SSM.
// x_ssm = xz[:, :, 0:128].  Outputs f32 (scan) + f16 (GEMMs).
// ---------------------------------------------------------------------------
__global__ void k_conv(const float* __restrict__ xz,
                       const float* __restrict__ conv_w, const float* __restrict__ conv_b,
                       float* __restrict__ xcf32, _Float16* __restrict__ xcf16, int reverse)
{
    int idx = blockIdx.x * 256 + threadIdx.x;
    if (idx >= B_SZ * D_MODEL) return;
    int b = idx >> 7, d = idx & 127;
    float w[4];
#pragma unroll
    for (int j = 0; j < 4; ++j) w[j] = conv_w[d * 4 + j];
    float bb = conv_b[d];

    float xv[N_WIN];
#pragma unroll
    for (int t = 0; t < N_WIN; ++t)
        xv[t] = xz[((size_t)(b * N_WIN + t)) * 256 + d];

#pragma unroll
    for (int t = 0; t < N_WIN; ++t) {
        float acc = bb;
#pragma unroll
        for (int j = 0; j < 4; ++j) {
            int s = reverse ? (t + 3 - j) : (t - 3 + j);
            if (s >= 0 && s < N_WIN) acc += w[j] * xv[s];
        }
        size_t o = (size_t)(b * N_WIN + t) * D_MODEL + d;
        xcf32[o] = acc;
        xcf16[o] = (_Float16)acc;
    }
}

// ---------------------------------------------------------------------------
// Selective-scan recurrence over T=12, fused with silu(z) gating.
// One thread per (b, d); 16 f32 states in registers.
// ---------------------------------------------------------------------------
__global__ __launch_bounds__(128) void k_scan(
    const float* __restrict__ dtb, const float* __restrict__ bc,
    const float* __restrict__ xcf32, const float* __restrict__ xz,
    const float* __restrict__ A_log, const float* __restrict__ Dp,
    _Float16* __restrict__ yz, int reverse)
{
    int b = blockIdx.x;
    int d = threadIdx.x;
    float an[D_STATE];
#pragma unroll
    for (int n = 0; n < D_STATE; ++n) an[n] = -expf(A_log[d * D_STATE + n]);
    float Dd = Dp[d];
    float h[D_STATE];
#pragma unroll
    for (int n = 0; n < D_STATE; ++n) h[n] = 0.0f;

    for (int s = 0; s < N_WIN; ++s) {
        int t = reverse ? (N_WIN - 1 - s) : s;
        size_t r = (size_t)(b * N_WIN + t);
        float dt = dtb[r * D_MODEL + d];
        float x  = xcf32[r * D_MODEL + d];
        float z  = xz[r * 256 + D_MODEL + d];
        float y = 0.0f;
#pragma unroll
        for (int n = 0; n < D_STATE; ++n) {
            float Bn = bc[r * 32 + n];
            float Cn = bc[r * 32 + D_STATE + n];
            h[n] = h[n] * expf(an[n] * dt) + dt * x * Bn;
            y += h[n] * Cn;
        }
        y += x * Dd;
        yz[r * D_MODEL + d] = (_Float16)(y * silu_f(z));
    }
}

// ---------------------------------------------------------------------------
// Final epilogue: hfinal = LN(h0 + merged) -> f16 (feeds head GEMM)
// ---------------------------------------------------------------------------
__global__ __launch_bounds__(256) void k_epi_final(
    const float* __restrict__ h0f32, const float* __restrict__ merged,
    const float* __restrict__ g, const float* __restrict__ bta,
    _Float16* __restrict__ hfinal)
{
    int row  = blockIdx.x * 8 + (threadIdx.x >> 5);
    int lane = threadIdx.x & 31;
    const float* hr = h0f32 + (size_t)row * D_MODEL;
    const float* mr = merged + (size_t)row * D_MODEL;

    float v[4], sum = 0.0f;
#pragma unroll
    for (int j = 0; j < 4; ++j) { v[j] = hr[lane + 32 * j] + mr[lane + 32 * j]; sum += v[j]; }
    sum = wave_sum32(sum);
    float mu = sum * (1.0f / 128.0f);
    float sq = 0.0f;
#pragma unroll
    for (int j = 0; j < 4; ++j) { float d = v[j] - mu; sq += d * d; }
    sq = wave_sum32(sq);
    float rstd = rsqrtf(sq * (1.0f / 128.0f) + 1e-5f);

#pragma unroll
    for (int j = 0; j < 4; ++j) {
        int c = lane + 32 * j;
        float y = (v[j] - mu) * rstd * g[c] + bta[c];
        hfinal[(size_t)row * D_MODEL + c] = (_Float16)y;
    }
}

// ---------------------------------------------------------------------------
// Host side
// ---------------------------------------------------------------------------
// input index map (setup_inputs dict order, params flattened in insertion order)
enum {
    IN_EMB = 0, IN_SCORES, IN_SITE, IN_HOURS,
    IN_INP_W, IN_INP_B, IN_LN_G, IN_LN_B,
    IN_SITE_EMB, IN_HOUR_EMB, IN_META_W, IN_META_B, IN_POS,
    F_INPROJ, F_CONVW, F_CONVB, F_DTW, F_DTB, F_ALOG, F_D, F_BW, F_CW, F_OUTP,
    B_INPROJ, B_CONVW, B_CONVB, B_DTW, B_DTB, B_ALOG, B_D, B_BW, B_CW, B_OUTP,
    IN_MERGE_W, IN_MERGE_B, IN_NORM_G, IN_NORM_B, IN_HEAD_W, IN_HEAD_B
};

extern "C" void kernel_launch(void* const* d_in, const int* in_sizes, int n_in,
                              void* d_out, int out_size, void* d_ws, size_t ws_size,
                              hipStream_t stream)
{
    (void)in_sizes; (void)n_in; (void)out_size; (void)ws_size;

    const float* emb    = (const float*)d_in[IN_EMB];
    const float* scores = (const float*)d_in[IN_SCORES];
    const int*   sites  = (const int*)d_in[IN_SITE];
    const int*   hours  = (const int*)d_in[IN_HOURS];

    // ---- workspace carve-out ----------------------------------------------
    char* wsp = (char*)d_ws;
    size_t off = 0;
    auto alloc = [&](size_t bytes) -> void* {
        void* p = wsp + off;
        off += (bytes + 255) & ~(size_t)255;
        return p;
    };
    _Float16* wt_inp   = (_Float16*)alloc((size_t)D_MODEL * K_IN_PAD * 2);   // [128][1792]
    _Float16* inprojT[2], *dtwT[2], *bcT[2], *outpT[2];
    for (int d = 0; d < 2; ++d) {
        inprojT[d] = (_Float16*)alloc((size_t)256 * 128 * 2);
        dtwT[d]    = (_Float16*)alloc((size_t)128 * 128 * 2);
        bcT[d]     = (_Float16*)alloc((size_t)32 * 128 * 2);
        outpT[d]   = (_Float16*)alloc((size_t)128 * 128 * 2);
    }
    _Float16* mergeT  = (_Float16*)alloc((size_t)128 * 256 * 2);
    _Float16* headT   = (_Float16*)alloc((size_t)N_CLS_PAD * 128 * 2);
    float*    meta_v  = (float*)alloc((size_t)B_SZ * D_MODEL * 4);
    float*    h_pre   = (float*)alloc((size_t)M_ROWS * D_MODEL * 4);   // reused as `merged`
    float*    h0f32   = (float*)alloc((size_t)M_ROWS * D_MODEL * 4);
    _Float16* h0f16   = (_Float16*)alloc((size_t)M_ROWS * D_MODEL * 2);
    float*    xzbuf   = (float*)alloc((size_t)M_ROWS * 256 * 4);
    float*    xcf32   = (float*)alloc((size_t)M_ROWS * D_MODEL * 4);
    _Float16* xcf16   = (_Float16*)alloc((size_t)M_ROWS * D_MODEL * 2);
    float*    dtbuf   = (float*)alloc((size_t)M_ROWS * D_MODEL * 4);
    float*    bcbuf   = (float*)alloc((size_t)M_ROWS * 32 * 4);
    _Float16* yzbuf   = (_Float16*)alloc((size_t)M_ROWS * D_MODEL * 2); // reused as hfinal
    _Float16* hcat    = (_Float16*)alloc((size_t)M_ROWS * 256 * 2);
    float*    merged  = h_pre;            // reuse (h_pre dead after epi_input)
    _Float16* hfinal  = yzbuf;            // reuse (yz dead after out_proj GEMMs)

    auto tpose = [&](const float* src, _Float16* dst, int K, int N, int Kp, int Np) {
        int total = Kp * Np;
        k_transpose_f16<<<(total + 255) / 256, 256, 0, stream>>>(src, dst, K, N, Kp, Np);
    };

    // ---- 1. weight prep ---------------------------------------------------
    tpose((const float*)d_in[IN_INP_W], wt_inp, D_IN_TOT, D_MODEL, K_IN_PAD, D_MODEL);
    const int IP[2] = {F_INPROJ, B_INPROJ}, DW[2] = {F_DTW, B_DTW};
    const int BW[2] = {F_BW, B_BW}, CW[2] = {F_CW, B_CW}, OP[2] = {F_OUTP, B_OUTP};
    for (int d = 0; d < 2; ++d) {
        tpose((const float*)d_in[IP[d]], inprojT[d], 128, 256, 128, 256);
        tpose((const float*)d_in[DW[d]], dtwT[d],    128, 128, 128, 128);
        tpose((const float*)d_in[BW[d]], bcT[d],           128, 16, 128, 16);
        tpose((const float*)d_in[CW[d]], bcT[d] + 16 * 128, 128, 16, 128, 16);
        tpose((const float*)d_in[OP[d]], outpT[d],   128, 128, 128, 128);
    }
    tpose((const float*)d_in[IN_MERGE_W], mergeT, 256, 128, 256, 128);
    tpose((const float*)d_in[IN_HEAD_W],  headT,  128, N_CLASSES, 128, N_CLS_PAD);

    // ---- 2. meta path -----------------------------------------------------
    k_meta<<<(B_SZ * D_MODEL + 255) / 256, 256, 0, stream>>>(
        sites, hours, (const float*)d_in[IN_SITE_EMB], (const float*)d_in[IN_HOUR_EMB],
        (const float*)d_in[IN_META_W], (const float*)d_in[IN_META_B], meta_v);

    // ---- 3. big input GEMM (concat fp32->f16 on the fly, K=1792) ----------
    k_gemm_wmma<8, 1, EPI_F32><<<dim3(M_ROWS / 64, 1), 128, 0, stream>>>(
        nullptr, emb, scores, wt_inp, (const float*)d_in[IN_INP_B],
        h_pre, nullptr, M_ROWS, K_IN_PAD, 0, K_IN_PAD, D_MODEL, D_MODEL);

    // ---- 4. LN + GELU + pos + meta ---------------------------------------
    k_epi_input<<<M_ROWS / 8, 256, 0, stream>>>(
        h_pre, (const float*)d_in[IN_LN_G], (const float*)d_in[IN_LN_B],
        (const float*)d_in[IN_POS], meta_v, h0f32, h0f16);

    // ---- 5. both SSM directions ------------------------------------------
    const int CVW[2] = {F_CONVW, B_CONVW}, CVB[2] = {F_CONVB, B_CONVB};
    const int DTB[2] = {F_DTB, B_DTB}, AL[2] = {F_ALOG, B_ALOG}, DD[2] = {F_D, B_D};
    for (int dir = 0; dir < 2; ++dir) {
        int rev = dir;  // dir 1 = time-reversed (bwd)
        // xz = h0 @ in_proj  (K=128, N=256)
        k_gemm_wmma<8, 0, EPI_F32><<<dim3(M_ROWS / 64, 2), 128, 0, stream>>>(
            h0f16, nullptr, nullptr, inprojT[dir], nullptr,
            xzbuf, nullptr, M_ROWS, 128, 128, 128, 256, 256);
        // depthwise causal conv
        k_conv<<<(B_SZ * D_MODEL + 255) / 256, 256, 0, stream>>>(
            xzbuf, (const float*)d_in[CVW[dir]], (const float*)d_in[CVB[dir]],
            xcf32, xcf16, rev);
        // dt = softplus(xconv @ dt_w + dt_b)
        k_gemm_wmma<8, 0, EPI_F32 | EPI_SOFTPLUS><<<dim3(M_ROWS / 64, 1), 128, 0, stream>>>(
            xcf16, nullptr, nullptr, dtwT[dir], (const float*)d_in[DTB[dir]],
            dtbuf, nullptr, M_ROWS, 128, 128, 128, 128, 128);
        // [B|C] = xconv @ [B_w|C_w]  (N=32)
        k_gemm_wmma<2, 0, EPI_F32><<<dim3(M_ROWS / 64, 1), 128, 0, stream>>>(
            xcf16, nullptr, nullptr, bcT[dir], nullptr,
            bcbuf, nullptr, M_ROWS, 128, 128, 128, 32, 32);
        // recurrence + silu(z) gate
        k_scan<<<B_SZ, 128, 0, stream>>>(
            dtbuf, bcbuf, xcf32, xzbuf,
            (const float*)d_in[AL[dir]], (const float*)d_in[DD[dir]], yzbuf, rev);
        // h_dir = yz @ out_proj -> f16 into hcat columns [dir*128, dir*128+128)
        k_gemm_wmma<8, 0, EPI_F16><<<dim3(M_ROWS / 64, 1), 128, 0, stream>>>(
            yzbuf, nullptr, nullptr, outpT[dir], nullptr,
            nullptr, hcat + dir * 128, M_ROWS, 128, 128, 128, 256, 128);
    }

    // ---- 6. merge GEMM (K=256, N=128) ------------------------------------
    k_gemm_wmma<8, 0, EPI_F32><<<dim3(M_ROWS / 64, 1), 128, 0, stream>>>(
        hcat, nullptr, nullptr, mergeT, (const float*)d_in[IN_MERGE_B],
        merged, nullptr, M_ROWS, 256, 256, 256, 128, 128);

    // ---- 7. residual + LN -------------------------------------------------
    k_epi_final<<<M_ROWS / 8, 256, 0, stream>>>(
        h0f32, merged, (const float*)d_in[IN_NORM_G], (const float*)d_in[IN_NORM_B], hfinal);

    // ---- 8. head GEMM (N padded 234->240, guarded store) ------------------
    k_gemm_wmma<5, 0, EPI_F32><<<dim3(M_ROWS / 64, 3), 128, 0, stream>>>(
        hfinal, nullptr, nullptr, headT, (const float*)d_in[IN_HEAD_B],
        (float*)d_out, nullptr, M_ROWS, 128, 128, 128, N_CLASSES, N_CLASSES);
}
